// DeepAR_34084860461707
// MI455X (gfx1250) — compile-verified, matrix operand order checked by reading, not run on hardware
//
#include <hip/hip_runtime.h>
#include <hip/hip_bf16.h>
#include <math.h>

typedef __attribute__((ext_vector_type(16))) __bf16 v16bf;
typedef __attribute__((ext_vector_type(8)))  float  v8f;
typedef __attribute__((ext_vector_type(4)))  float  f32x4;

#define STEPS 119
#define NROWS_TOTAL 32768   // B*N = 64*512
#define K0 352              // 67 (embed+cov) + 256 (h0), padded to mult of 32
#define K1 512              // 256 (h0) + 256 (h1)
#define MROWS 32            // rows per workgroup
#define PLANE (64L*119L*512L)

__device__ __forceinline__ float sigm(float x) {
  return 1.0f / (1.0f + __expf(-x));
}

// One K-sweep: acc[g*2+j] accumulates the 16x16 output tile for gate g,
// sub-tile j of this wave's 32-wide h slice. A comes from LDS (bf16), B from
// global (L2-resident packed bf16 weights, row-major [1024][ldw]).
// K-loop is kept rolled (#pragma unroll 1) so only ONE chunk's B fragments
// (64 VGPRs) are live at a time -> no scratch spills.
__device__ __forceinline__ void gemm_tiles(const __bf16* __restrict__ sA, int lda,
                                           const __bf16* __restrict__ W, int ldw,
                                           int nChunks, int mBase, int hs,
                                           int lane, v8f acc[8]) {
  const int khalf = (lane >> 4) & 1;
  const int l16   = lane & 15;
  const __bf16* aRow  = sA + (mBase + l16) * lda;
  const __bf16* wBase = W + (size_t)(hs + l16) * ldw;
#pragma unroll 1
  for (int kc = 0; kc < nChunks; ++kc) {
    const int kb = kc * 32 + khalf * 8;
    union { f32x4 q[2]; v16bf v; } ua;
    union { f32x4 q[2]; v16bf v; } ub[8];
    const f32x4* pa = (const f32x4*)(aRow + kb);
    ua.q[0] = pa[0];
    ua.q[1] = pa[2];          // +32B = +16 bf16 (second K half-group)
    // issue all 16 weight loads for this chunk before any WMMA
#pragma unroll
    for (int t = 0; t < 8; ++t) {
      const int g = t >> 1, j = t & 1;
      const f32x4* pb = (const f32x4*)(wBase + (size_t)(g * 256 + j * 16) * ldw + kb);
      ub[t].q[0] = pb[0];
      ub[t].q[1] = pb[2];
    }
#pragma unroll
    for (int t = 0; t < 8; ++t) {
      acc[t] = __builtin_amdgcn_wmma_f32_16x16x32_bf16(
          false, ua.v, false, ub[t].v, (short)0, acc[t], false, false);
    }
  }
}

__global__ void deepar_prep(const float* __restrict__ W_ih0, const float* __restrict__ W_hh0,
                            const float* __restrict__ b_ih0, const float* __restrict__ b_hh0,
                            const float* __restrict__ W_ih1, const float* __restrict__ W_hh1,
                            const float* __restrict__ b_ih1, const float* __restrict__ b_hh1,
                            __bf16* __restrict__ wc0, __bf16* __restrict__ wc1,
                            float* __restrict__ bias0, float* __restrict__ bias1) {
  int i = blockIdx.x * blockDim.x + threadIdx.x;
  if (i < 1024 * K0) {                     // wc0[n][k] = [W_ih0 | W_hh0 | 0pad]
    int n = i / K0, k = i % K0;
    float v = 0.0f;
    if (k < 67)       v = W_ih0[n * 67 + k];
    else if (k < 323) v = W_hh0[n * 256 + (k - 67)];
    wc0[i] = (__bf16)v;
  }
  int j = i - 1024 * K0;
  if (j >= 0 && j < 1024 * K1) {           // wc1[n][k] = [W_ih1 | W_hh1]
    int n = j / K1, k = j % K1;
    float v = (k < 256) ? W_ih1[n * 256 + k] : W_hh1[n * 256 + (k - 256)];
    wc1[j] = (__bf16)v;
  }
  int m = i - (1024 * K0 + 1024 * K1);
  if (m >= 0 && m < 1024)  bias0[m] = b_ih0[m] + b_hh0[m];
  int m2 = m - 1024;
  if (m2 >= 0 && m2 < 1024) bias1[m2] = b_ih1[m2] + b_hh1[m2];
}

__global__ void deepar_reals(const float* __restrict__ hist, const float* __restrict__ fut,
                             float* __restrict__ reals) {
  long i = blockIdx.x * (long)blockDim.x + threadIdx.x;
  if (i >= PLANE) return;
  int n = (int)(i % 512);
  long t2 = i / 512;
  int t = (int)(t2 % STEPS);
  int b = (int)(t2 / STEPS);
  int tt = t + 1;   // reals = z at times 1..119
  float z = (tt < 96) ? hist[(((long)b * 96 + tt) * 512 + n) * 4]
                      : fut [(((long)b * 24 + (tt - 96)) * 512 + n) * 4];
  reals[i] = z;
}

__global__ __launch_bounds__(512, 1)
void deepar_main(const float* __restrict__ hist, const float* __restrict__ fut,
                 const float* __restrict__ w_embed, const float* __restrict__ b_embed,
                 const __bf16* __restrict__ wc0, const __bf16* __restrict__ wc1,
                 const float* __restrict__ bias0, const float* __restrict__ bias1,
                 const float* __restrict__ w_mu, const float* __restrict__ b_mu,
                 const float* __restrict__ w_sg, const float* __restrict__ b_sg,
                 float* __restrict__ out_preds, float* __restrict__ out_mus,
                 float* __restrict__ out_sgs) {
  __shared__ __bf16 sA0[MROWS * K0];   // [x(67) | h0(256) | pad] bf16
  __shared__ __bf16 sA1[MROWS * K1];   // [h0(256) | h1(256)] bf16
  __shared__ float  sMu[MROWS];
  __shared__ float  sSg[MROWS];

  const int tid   = threadIdx.x;
  const int lane  = tid & 31;
  const int wave  = tid >> 5;
  const int waveM = wave >> 3;          // 0..1 : which 16-row tile
  const int hs    = (wave & 7) * 32;    // this wave's 32-wide h slice
  const int l16   = lane & 15;
  const int khalf = (lane >> 4) & 1;
  const int rowBase = blockIdx.x * MROWS;

  for (int i = tid; i < MROWS * K0; i += 512) sA0[i] = (__bf16)0.0f;
  for (int i = tid; i < MROWS * K1; i += 512) sA1[i] = (__bf16)0.0f;
  if (tid < MROWS) { sMu[tid] = 0.0f; sSg[tid] = 0.0f; }

  v8f c0[2] = {};                       // cell state lives in VGPRs, C-frag layout
  v8f c1[2] = {};
  __syncthreads();

#pragma unroll 1
  for (int s = 0; s < STEPS; ++s) {
    // ---- stage x_t = [embed(z_s), cov_{s+1}] into sA0[:,0:67] ----
    for (int idx = tid; idx < MROWS * 67; idx += 512) {
      int r = idx / 67, col = idx % 67;
      int row = rowBase + r;
      int b = row >> 9, n = row & 511;
      float v;
      if (col < 64) {
        float z = (s < 96) ? hist[(((long)b * 96 + s) * 512 + n) * 4]
                           : fut [(((long)b * 24 + (s - 96)) * 512 + n) * 4];
        v = z * w_embed[col] + b_embed[col];
      } else {
        int c = col - 64 + 1;
        int t1 = s + 1;
        v = (t1 < 96) ? hist[(((long)b * 96 + t1) * 512 + n) * 4 + c]
                      : fut [(((long)b * 24 + (t1 - 96)) * 512 + n) * 4 + c];
      }
      sA0[r * K0 + col] = (__bf16)v;
    }
    __syncthreads();

    // ---- layer 0 GEMM: g0 = [x|h0] @ Wcat0^T ----
    v8f acc[8] = {};
    gemm_tiles(sA0, K0, wc0, K0, K0 / 32, waveM * 16, hs, lane, acc);
    __syncthreads();   // all A0 reads done before h0 is overwritten

    // ---- layer 0 cell update, entirely in registers ----
#pragma unroll
    for (int j = 0; j < 2; ++j) {
      const int hcol = hs + j * 16 + l16;
      const float bI = bias0[0 * 256 + hcol];
      const float bF = bias0[1 * 256 + hcol];
      const float bG = bias0[2 * 256 + hcol];
      const float bO = bias0[3 * 256 + hcol];
#pragma unroll
      for (int v = 0; v < 8; ++v) {
        float gi = sigm(acc[0 * 2 + j][v] + bI);
        float gf = sigm(acc[1 * 2 + j][v] + bF);
        float gg = tanhf(acc[2 * 2 + j][v] + bG);
        float go = sigm(acc[3 * 2 + j][v] + bO);
        float cn = gf * c0[j][v] + gi * gg;
        c0[j][v] = cn;
        float h = go * tanhf(cn);
        int rloc = waveM * 16 + v + 8 * khalf;   // C-frag row mapping
        __bf16 hb = (__bf16)h;
        sA0[rloc * K0 + 67 + hcol] = hb;         // feeds next step layer0
        sA1[rloc * K1 + hcol]      = hb;         // feeds this step layer1
      }
    }
    __syncthreads();

    // ---- layer 1 GEMM: g1 = [h0|h1] @ Wcat1^T ----
    v8f acc1[8] = {};
    gemm_tiles(sA1, K1, wc1, K1, K1 / 32, waveM * 16, hs, lane, acc1);
    __syncthreads();   // all A1 reads done before h1 is overwritten

    // ---- layer 1 cell update + in-register head partial dots ----
    float vm[8], vs[8];
#pragma unroll
    for (int v = 0; v < 8; ++v) { vm[v] = 0.0f; vs[v] = 0.0f; }
#pragma unroll
    for (int j = 0; j < 2; ++j) {
      const int hcol = hs + j * 16 + l16;
      const float bI = bias1[0 * 256 + hcol];
      const float bF = bias1[1 * 256 + hcol];
      const float bG = bias1[2 * 256 + hcol];
      const float bO = bias1[3 * 256 + hcol];
      const float wm = w_mu[hcol];
      const float wsg = w_sg[hcol];
#pragma unroll
      for (int v = 0; v < 8; ++v) {
        float gi = sigm(acc1[0 * 2 + j][v] + bI);
        float gf = sigm(acc1[1 * 2 + j][v] + bF);
        float gg = tanhf(acc1[2 * 2 + j][v] + bG);
        float go = sigm(acc1[3 * 2 + j][v] + bO);
        float cn = gf * c1[j][v] + gi * gg;
        c1[j][v] = cn;
        float h = go * tanhf(cn);
        int rloc = waveM * 16 + v + 8 * khalf;
        sA1[rloc * K1 + 256 + hcol] = (__bf16)h;  // feeds next step layer1
        float hr = fmaxf(h, 0.0f);
        vm[v] += hr * wm;
        vs[v] += hr * wsg;
      }
    }
    // reduce across the 16 column-lanes of each half-wave, then LDS atomic add
#pragma unroll
    for (int v = 0; v < 8; ++v) {
      float m = vm[v], g = vs[v];
      for (int off = 8; off > 0; off >>= 1) {
        m += __shfl_xor(m, off, 32);
        g += __shfl_xor(g, off, 32);
      }
      if (l16 == 0) {
        int rloc = waveM * 16 + v + 8 * khalf;
        atomicAdd(&sMu[rloc], m);
        atomicAdd(&sSg[rloc], g);
      }
    }
    __syncthreads();

    // ---- head finalize: mu, sigma, pred ----
    if (tid < MROWS) {
      int row = rowBase + tid;
      int b = row >> 9, n = row & 511;
      float mu = sMu[tid] + b_mu[0];
      float x  = sSg[tid] + b_sg[0];
      float sp = (x > 20.0f) ? x : log1pf(__expf(x));
      float sg = sp + 1e-6f;
      // hash-based Box-Muller for eps (reference JAX PRNG not reproducible)
      unsigned u = (unsigned)(s * NROWS_TOTAL + row) * 2654435761u + 0x9E3779B9u;
      u ^= u >> 16; u *= 0x85EBCA6Bu; u ^= u >> 13; u *= 0xC2B2AE35u; u ^= u >> 16;
      unsigned u2 = u * 747796405u + 2891336453u; u2 ^= u2 >> 17;
      float f1 = (float)(u  & 0xFFFFFF) * (1.0f / 16777216.0f) + 1e-7f;
      float f2 = (float)(u2 & 0xFFFFFF) * (1.0f / 16777216.0f);
      float eps = sqrtf(-2.0f * __logf(f1)) * __cosf(6.2831853f * f2);
      float pred = mu + sg * eps;
      long o = ((long)b * STEPS + s) * 512 + n;
      out_preds[o] = pred;
      out_mus[o]   = mu;
      out_sgs[o]   = sg;
      sMu[tid] = 0.0f;   // reset accumulators for next step
      sSg[tid] = 0.0f;
    }
    __syncthreads();
  }
}

extern "C" void kernel_launch(void* const* d_in, const int* in_sizes, int n_in,
                              void* d_out, int out_size, void* d_ws, size_t ws_size,
                              hipStream_t stream) {
  const float* hist    = (const float*)d_in[0];
  const float* fut     = (const float*)d_in[1];
  // d_in[2] = train flag (int), unused
  const float* w_embed = (const float*)d_in[3];
  const float* b_embed = (const float*)d_in[4];
  const float* W_ih0   = (const float*)d_in[5];
  const float* W_hh0   = (const float*)d_in[6];
  const float* b_ih0   = (const float*)d_in[7];
  const float* b_hh0   = (const float*)d_in[8];
  const float* W_ih1   = (const float*)d_in[9];
  const float* W_hh1   = (const float*)d_in[10];
  const float* b_ih1   = (const float*)d_in[11];
  const float* b_hh1   = (const float*)d_in[12];
  const float* W_mu    = (const float*)d_in[13];
  const float* b_mu    = (const float*)d_in[14];
  const float* W_sg    = (const float*)d_in[15];
  const float* b_sg    = (const float*)d_in[16];

  char* ws = (char*)d_ws;
  __bf16* wc0  = (__bf16*)ws;                                   // 1024*352*2 B
  __bf16* wc1  = (__bf16*)(ws + 1024 * K0 * 2);                 // 1024*512*2 B
  float*  bias0 = (float*)(ws + 1024 * K0 * 2 + 1024 * K1 * 2); // 4 KB
  float*  bias1 = bias0 + 1024;

  float* out   = (float*)d_out;
  float* preds = out;
  float* reals = out + PLANE;
  float* mus   = out + 2 * PLANE;
  float* sgs   = out + 3 * PLANE;

  int prep_total = 1024 * K0 + 1024 * K1 + 2048;
  deepar_prep<<<(prep_total + 255) / 256, 256, 0, stream>>>(
      W_ih0, W_hh0, b_ih0, b_hh0, W_ih1, W_hh1, b_ih1, b_hh1,
      wc0, wc1, bias0, bias1);

  deepar_reals<<<(int)((PLANE + 255) / 256), 256, 0, stream>>>(hist, fut, reals);

  deepar_main<<<NROWS_TOTAL / MROWS, 512, 0, stream>>>(
      hist, fut, w_embed, b_embed, wc0, wc1, bias0, bias1,
      W_mu, b_mu, W_sg, b_sg, preds, mus, sgs);
}
